// Head_10926396801290
// MI455X (gfx1250) — matrix operand order repaired and checked by reading, not compiled, not run on hardware
//
#include <hip/hip_runtime.h>
#include <hip/hip_bf16.h>

typedef _Float16 half_t;
typedef __attribute__((ext_vector_type(16))) _Float16 v16h;
typedef __attribute__((ext_vector_type(8)))  float    v8f;

#define B_BATCH 512
#define T_SEQ   256
#define C_EMB   512
#define HSZ     64

union V16U {
    v16h  v;
    uint4 q[2];
    half_t h[16];
};

union H4U {
    half_t h[4];
    uint2  u;
};

// ---------------------------------------------------------------------------
// Kernel 1: fused QKV projection.
//   One block per batch (reads its x slice exactly once, via LDS fp16 stage).
//   12 waves = 3 matrices x 4 h-tiles; each wave preloads its W slice into
//   B-layout registers (v16h per 32-k chunk) and runs 16 WMMAs per row tile.
//   Outputs: Qh, Kh row-major [b][t][64] fp16; V transposed Vt [b][64][t] fp16
//   so that the attention kernel's B operands are k-contiguous.
// ---------------------------------------------------------------------------
__global__ __launch_bounds__(384) void qkv_proj_kernel(
    const float* __restrict__ x,  const float* __restrict__ Wq,
    const float* __restrict__ Wk, const float* __restrict__ Wv,
    half_t* __restrict__ qh, half_t* __restrict__ kh, half_t* __restrict__ vt)
{
    __shared__ __attribute__((aligned(16))) half_t xs[16][C_EMB + 8];

    const int b    = blockIdx.x;
    const int tid  = threadIdx.x;
    const int wave = tid >> 5;
    const int lane = tid & 31;
    const int ln   = lane & 15;   // column / row-in-tile index
    const int g    = lane >> 4;   // half-wave group
    const int mi   = wave >> 2;   // 0=Q 1=K 2=V
    const int ht   = wave & 3;    // 16-wide h tile

    const float* W = (mi == 0) ? Wq : (mi == 1) ? Wk : Wv;

    // Preload W in B-layout: element e of chunk c holds k = 32c + 16g + e.
    v16h WB[16];
#pragma unroll
    for (int c = 0; c < 16; ++c) {
        V16U u;
#pragma unroll
        for (int e = 0; e < 16; ++e) {
            const int k = 32 * c + 16 * g + e;
            u.h[e] = (half_t)W[k * HSZ + 16 * ht + ln];
        }
        WB[c] = u.v;
    }

    for (int it = 0; it < 16; ++it) {
        __syncthreads();                       // protect LDS reuse
        // Stage 16x512 fp32 x tile -> fp16 LDS, vectorized:
        // 2048 float4 global loads (b128), 8-byte LDS stores (b64).
        for (int idx = tid; idx < (16 * C_EMB) / 4; idx += 384) {
            const int r  = idx >> 7;           // 128 float4 per row
            const int c4 = idx & 127;
            const float4 f =
                *(const float4*)&x[((size_t)b * T_SEQ + it * 16 + r) * C_EMB + c4 * 4];
            H4U h;
            h.h[0] = (half_t)f.x;
            h.h[1] = (half_t)f.y;
            h.h[2] = (half_t)f.z;
            h.h[3] = (half_t)f.w;
            *(uint2*)&xs[r][c4 * 4] = h.u;
        }
        __syncthreads();

        v8f acc = {};
#pragma unroll
        for (int c = 0; c < 16; ++c) {
            // A layout: lane row m=ln; k runs [32c+16g, +16)
            V16U a;
            a.q[0] = *(const uint4*)&xs[ln][32 * c + 16 * g];
            a.q[1] = *(const uint4*)&xs[ln][32 * c + 16 * g + 8];
            acc = __builtin_amdgcn_wmma_f32_16x16x32_f16(
                false, a.v, false, WB[c], (short)0, acc, false, false);
        }

        // C layout: element r -> row m = r + 8g, lane -> column h
#pragma unroll
        for (int r = 0; r < 8; ++r) {
            const int trow = it * 16 + r + 8 * g;
            const int h    = 16 * ht + ln;
            const half_t v = (half_t)acc[r];
            if (mi == 0)      qh[((size_t)b * T_SEQ + trow) * HSZ + h] = v;
            else if (mi == 1) kh[((size_t)b * T_SEQ + trow) * HSZ + h] = v;
            else              vt[((size_t)b * HSZ + h) * T_SEQ + trow] = v;
        }
    }
}

// ---------------------------------------------------------------------------
// Kernel 2: causal attention. 8 waves/block, each wave owns one 16-row tile;
// full 16x256 score block lives in 16 v8f accumulators (no online softmax
// needed at T=256). P is re-laid-out C->A through a per-wave LDS buffer.
// ---------------------------------------------------------------------------
__global__ __launch_bounds__(256) void attn_kernel(
    const half_t* __restrict__ qh, const half_t* __restrict__ kh,
    const half_t* __restrict__ vt, float* __restrict__ out)
{
    __shared__ __attribute__((aligned(16))) half_t plds[8][16][40];

    const int b    = blockIdx.x >> 1;
    const int tid  = threadIdx.x;
    const int wave = tid >> 5;
    const int lane = tid & 31;
    const int ln   = lane & 15;
    const int g    = lane >> 4;
    const int it   = (blockIdx.x & 1) * 8 + wave;   // 16-row tile index in T

    const float scale = 0.04419417382415922f;       // 512^-0.5 (C = n_embd)
    const float LOG2E = 1.4426950408889634f;

    // Q tile in A layout (K dim = head = 64 -> two 32-k chunks)
    const half_t* qrow = qh + ((size_t)b * T_SEQ + it * 16 + ln) * HSZ;
    v16h AQ[2];
#pragma unroll
    for (int c = 0; c < 2; ++c) {
        V16U u;
        u.q[0] = *(const uint4*)(qrow + 32 * c + 16 * g);
        u.q[1] = *(const uint4*)(qrow + 32 * c + 16 * g + 8);
        AQ[c] = u.v;
    }

    // S = Q K^T, causal tiles only (j <= it)
    v8f S[16] = {};
#pragma unroll
    for (int j = 0; j < 16; ++j) {
        if (j <= it) {
            const half_t* krow = kh + ((size_t)b * T_SEQ + j * 16 + ln) * HSZ;
#pragma unroll
            for (int c = 0; c < 2; ++c) {
                // B layout: lane column n = ln (a t index), k = 32c + 16g + e,
                // contiguous in h since Kh is [t][64] row-major.
                V16U u;
                u.q[0] = *(const uint4*)(krow + 32 * c + 16 * g);
                u.q[1] = *(const uint4*)(krow + 32 * c + 16 * g + 8);
                S[j] = __builtin_amdgcn_wmma_f32_16x16x32_f16(
                    false, AQ[c], false, u.v, (short)0, S[j], false, false);
            }
        }
    }

    // scale + causal mask + row max
    float mx[8];
#pragma unroll
    for (int r = 0; r < 8; ++r) mx[r] = -1e30f;
#pragma unroll
    for (int j = 0; j < 16; ++j) {
        if (j <= it) {
#pragma unroll
            for (int r = 0; r < 8; ++r) {
                const int n = j * 16 + ln;
                const int m = it * 16 + r + 8 * g;
                float v = S[j][r] * scale;
                if (n > m) v = -1e30f;
                S[j][r] = v;
                mx[r] = fmaxf(mx[r], v);
            }
        }
    }
#pragma unroll
    for (int r = 0; r < 8; ++r) {
#pragma unroll
        for (int off = 1; off < 16; off <<= 1)
            mx[r] = fmaxf(mx[r], __shfl_xor(mx[r], off, 32));
    }

    // exp + row sum (normalize at the end; P·V is linear)
    float rs[8];
#pragma unroll
    for (int r = 0; r < 8; ++r) rs[r] = 0.0f;
#pragma unroll
    for (int j = 0; j < 16; ++j) {
        if (j <= it) {
#pragma unroll
            for (int r = 0; r < 8; ++r) {
                const float p = exp2f((S[j][r] - mx[r]) * LOG2E);
                S[j][r] = p;
                rs[r] += p;
            }
        }
    }
#pragma unroll
    for (int r = 0; r < 8; ++r) {
#pragma unroll
        for (int off = 1; off < 16; off <<= 1)
            rs[r] += __shfl_xor(rs[r], off, 32);
    }

    // O = P V via WMMA; P goes C-layout -> LDS (row-major 16x32) -> A-layout
    v8f O[4] = {};
#pragma unroll
    for (int jc = 0; jc < 8; ++jc) {
        if (2 * jc <= it) {
#pragma unroll
            for (int jj = 0; jj < 2; ++jj) {
                const int j = 2 * jc + jj;
#pragma unroll
                for (int r = 0; r < 8; ++r) {
                    const float p = (j <= it) ? S[j][r] : 0.0f;
                    plds[wave][r + 8 * g][jj * 16 + ln] = (half_t)p;
                }
            }
            // same-wave LDS RAW: compiler inserts the DScnt wait
            V16U a;
            a.q[0] = *(const uint4*)&plds[wave][ln][16 * g];
            a.q[1] = *(const uint4*)&plds[wave][ln][16 * g + 8];
#pragma unroll
            for (int nt = 0; nt < 4; ++nt) {
                // B operand from Vt [b][h][t]: lane column n = h, k = s
                const half_t* vrow =
                    vt + ((size_t)b * HSZ + 16 * nt + ln) * T_SEQ + 32 * jc;
                V16U u;
                u.q[0] = *(const uint4*)(vrow + 16 * g);
                u.q[1] = *(const uint4*)(vrow + 16 * g + 8);
                O[nt] = __builtin_amdgcn_wmma_f32_16x16x32_f16(
                    false, a.v, false, u.v, (short)0, O[nt], false, false);
            }
        }
    }

    // normalize by row sum, store fp32
#pragma unroll
    for (int nt = 0; nt < 4; ++nt) {
#pragma unroll
        for (int r = 0; r < 8; ++r) {
            out[((size_t)b * T_SEQ + it * 16 + r + 8 * g) * HSZ + 16 * nt + ln] =
                O[nt][r] / rs[r];
        }
    }
}

extern "C" void kernel_launch(void* const* d_in, const int* in_sizes, int n_in,
                              void* d_out, int out_size, void* d_ws, size_t ws_size,
                              hipStream_t stream) {
    const float* x  = (const float*)d_in[0];
    const float* Wq = (const float*)d_in[1];
    const float* Wk = (const float*)d_in[2];
    const float* Wv = (const float*)d_in[3];
    float* out = (float*)d_out;

    // workspace: Qh, Kh [B][T][64] fp16 ; Vt [B][64][T] fp16  (48 MB total)
    half_t* qh = (half_t*)d_ws;
    half_t* kh = qh + (size_t)B_BATCH * T_SEQ * HSZ;
    half_t* vt = kh + (size_t)B_BATCH * T_SEQ * HSZ;

    qkv_proj_kernel<<<B_BATCH, 384, 0, stream>>>(x, Wq, Wk, Wv, qh, kh, vt);
    attn_kernel<<<B_BATCH * 2, 256, 0, stream>>>(qh, kh, vt, out);
}
